// HRRAdaptedAttention_82420422410940
// MI455X (gfx1250) — matrix operand order repaired and checked by reading, not compiled, not run on hardware
//
#include <hip/hip_runtime.h>
#include <hip/hip_bf16.h>

// ---------------------------------------------------------------------------
// Problem constants (from reference setup_inputs)
// ---------------------------------------------------------------------------
#define BB   2
#define SS   4096
#define DD   2048
#define MM   (BB * SS)          // 8192 rows for the GEMMs
#define LOGD 11

typedef __attribute__((ext_vector_type(16))) __bf16 bf16x16;
typedef __attribute__((ext_vector_type(8)))  __bf16 bf16x8;
typedef __attribute__((ext_vector_type(8)))  float  floatx8;
typedef __attribute__((ext_vector_type(4)))  unsigned int uint32x4;
typedef __attribute__((ext_vector_type(8)))  int  int32x8;
typedef __attribute__((ext_vector_type(4)))  int  int32x4;

struct c32 { float x, y; };

#if defined(__has_builtin)
#if __has_builtin(__builtin_amdgcn_tensor_load_to_lds)
#define HRR_HAVE_TDM 1
#endif
#endif

// ---------------------------------------------------------------------------
// f32 -> bf16 (round to nearest even)
// ---------------------------------------------------------------------------
__device__ __forceinline__ unsigned short f2bf(float f) {
    unsigned u = __float_as_uint(f);
    unsigned r = ((u >> 16) & 1u) + 0x7FFFu;
    return (unsigned short)((u + r) >> 16);
}

__global__ __launch_bounds__(256)
void hrr_convert_bf16(const float* __restrict__ hs,
                      const float* __restrict__ wq,
                      const float* __restrict__ wk,
                      const float* __restrict__ wv,
                      unsigned short* __restrict__ hsBf,
                      unsigned short* __restrict__ wBf) {
    const size_t nHs = (size_t)MM * DD;
    const size_t nW  = (size_t)DD * DD;
    const size_t tot = nHs + 3 * nW;
    size_t stride = (size_t)gridDim.x * blockDim.x;
    for (size_t idx = (size_t)blockIdx.x * blockDim.x + threadIdx.x; idx < tot; idx += stride) {
        if (idx < nHs) {
            hsBf[idx] = f2bf(hs[idx]);
        } else {
            size_t j = idx - nHs;
            size_t w = j / nW, r = j % nW;
            const float* src = (w == 0) ? wq : (w == 1) ? wk : wv;
            wBf[j] = f2bf(src[r]);
        }
    }
}

// ---------------------------------------------------------------------------
// TDM: async DMA of a 2D bf16 tile (tileRows x 32 K-elements) into LDS.
// D# layout per CDNA5 ISA 8.3/8.4: count=1, type=2, data_size=1 (2B),
// tensor_dim0 = DD (K), tensor_dim0_stride = DD, groups 2/3 zero (2-D).
// ---------------------------------------------------------------------------
#ifdef HRR_HAVE_TDM
__device__ __forceinline__ void tdm_load_2d(unsigned ldsOff, unsigned long long ga,
                                            unsigned tileRows, unsigned rowsTot) {
    uint32x4 g0;
    g0[0] = 1u;                                              // count=1, user mode
    g0[1] = ldsOff;                                          // lds_addr (bytes)
    g0[2] = (unsigned)(ga & 0xFFFFFFFFull);                  // global_addr[31:0]
    g0[3] = (unsigned)((ga >> 32) & 0x1FFFFFFull)            // global_addr[56:32]
          | (2u << 30);                                      // type=2 (image)
    int32x8 g1;
    g1[0] = (int)(1u << 16);                                 // data_size=1 (2 bytes)
    g1[1] = (int)(((unsigned)DD & 0xFFFFu) << 16);           // tensor_dim0[15:0]
    g1[2] = (int)((((unsigned)DD >> 16) & 0xFFFFu)
          | ((rowsTot & 0xFFFFu) << 16));                    // dim0 hi | tensor_dim1 lo
    g1[3] = (int)(((rowsTot >> 16) & 0xFFFFu)
          | (32u << 16));                                    // tensor_dim1 hi | tile_dim0=32
    g1[4] = (int)(tileRows & 0xFFFFu);                       // tile_dim1 (tile_dim2=0)
    g1[5] = (int)DD;                                         // tensor_dim0_stride[31:0]
    g1[6] = 0;                                               // stride hi | dim1_stride lo
    g1[7] = 0;
    int32x4 gz = {0, 0, 0, 0};
#if defined(__clang_major__) && (__clang_major__ >= 23)
    int32x8 gz8 = {0, 0, 0, 0, 0, 0, 0, 0};
    __builtin_amdgcn_tensor_load_to_lds(g0, g1, gz, gz, gz8, 0);
#else
    __builtin_amdgcn_tensor_load_to_lds(g0, g1, gz, gz, 0);
#endif
}
#endif

// ---------------------------------------------------------------------------
// bf16 WMMA GEMM:  out[z][m][n] = sum_k hs[m][k] * W_z[n][k] + bias_z[n]
// block = 256 threads = 8 waves; wave tile = 16(M) x 64(N); block tile 32x256
// A tile (32x32) and B tile (256x32) staged in LDS by the Tensor Data Mover,
// double-buffered, synchronized with s_wait_tensorcnt + barriers.
// grid  = (DD/256, MM/32, 3)
// ---------------------------------------------------------------------------
__device__ __forceinline__ bf16x16 load_frag16(const unsigned short* p, int off2) {
    bf16x8 lo = *(const bf16x8*)(p);
    bf16x8 hi = *(const bf16x8*)(p + off2);
    return __builtin_shufflevector(lo, hi, 0,1,2,3,4,5,6,7,8,9,10,11,12,13,14,15);
}

// low 32 bits of the generic address of an LDS pointer == LDS byte offset
__device__ __forceinline__ unsigned lds_off(const void* p) {
    return (unsigned)(unsigned long long)p;
}

__global__ __launch_bounds__(256)
void hrr_gemm_qkv(const unsigned short* __restrict__ hsBf,
                  const unsigned short* __restrict__ wBf,
                  const float* __restrict__ b0,
                  const float* __restrict__ b1,
                  const float* __restrict__ b2,
                  float* __restrict__ qkv) {
    const int Ktot = DD, Ntot = DD;
    const int z = blockIdx.z;
    const unsigned short* w    = wBf + (size_t)z * Ntot * Ktot;
    const float*          bias = (z == 0) ? b0 : (z == 1) ? b1 : b2;
    float*                out  = qkv + (size_t)z * MM * Ntot;

    const int lane  = threadIdx.x & 31;
    const int wave  = threadIdx.x >> 5;
    const int waveM = wave >> 2;          // 0..1
    const int waveN = wave & 3;           // 0..3
    const int mBlk  = blockIdx.y * 32;
    const int nBlk  = blockIdx.x * 256;
    const int mBase = mBlk + waveM * 16;
    const int nBase = nBlk + waveN * 64;

    const int laneLo = lane & 15;
    const int laneHi = lane >> 4;

    floatx8 acc0 = {0.f,0.f,0.f,0.f,0.f,0.f,0.f,0.f};
    floatx8 acc1 = acc0, acc2 = acc0, acc3 = acc0;

#ifdef HRR_HAVE_TDM
    __shared__ unsigned short aT[2][32 * 32];    //  2 x  2 KB
    __shared__ unsigned short bT[2][256 * 32];   //  2 x 16 KB

    const unsigned long long gaA0 = (unsigned long long)(hsBf + (size_t)mBlk * Ktot);
    const unsigned long long gaB0 = (unsigned long long)(w    + (size_t)nBlk * Ktot);
    const unsigned aOff0 = lds_off(&aT[0][0]), aOff1 = lds_off(&aT[1][0]);
    const unsigned bOff0 = lds_off(&bT[0][0]), bOff1 = lds_off(&bT[1][0]);

    // wave-uniform lead test (tensor ops ignore EXEC -> must be a scalar branch)
    const bool tdmLead = (__builtin_amdgcn_readfirstlane((int)threadIdx.x) < 32);

    // per-lane fragment bases inside the LDS tiles (contiguous-K per-lane layouts, k = 16*laneHi + s in both A and B)
    const int aFragOff = (waveM * 16 + laneLo) * 32 + laneHi * 16;  // A: 16 contiguous K
    const int bFragOff0 = (waveN * 64 +  0 + laneLo) * 32 + laneHi * 16;  // B: 16 contiguous K
    const int bFragOff1 = (waveN * 64 + 16 + laneLo) * 32 + laneHi * 16;
    const int bFragOff2 = (waveN * 64 + 32 + laneLo) * 32 + laneHi * 16;
    const int bFragOff3 = (waveN * 64 + 48 + laneLo) * 32 + laneHi * 16;

    if (tdmLead) {
        tdm_load_2d(aOff0, gaA0, 32u,  (unsigned)MM);
        tdm_load_2d(bOff0, gaB0, 256u, (unsigned)DD);
    }

    for (int kk = 0; kk < Ktot; kk += 32) {
        const int cur = (kk >> 5) & 1;
        if (tdmLead) {
            if (kk + 32 < Ktot) {
                tdm_load_2d(cur ? aOff0 : aOff1, gaA0 + (size_t)(kk + 32) * 2, 32u,  (unsigned)MM);
                tdm_load_2d(cur ? bOff0 : bOff1, gaB0 + (size_t)(kk + 32) * 2, 256u, (unsigned)DD);
                __builtin_amdgcn_s_wait_tensorcnt(2);   // current pair landed
            } else {
                __builtin_amdgcn_s_wait_tensorcnt(0);
            }
        }
        __syncthreads();

        bf16x16 a = load_frag16(&aT[cur][aFragOff], 8);
        bf16x16 f0 = load_frag16(&bT[cur][bFragOff0], 8);
        acc0 = __builtin_amdgcn_wmma_f32_16x16x32_bf16(false, a, false, f0, (short)0, acc0, false, false);
        bf16x16 f1 = load_frag16(&bT[cur][bFragOff1], 8);
        acc1 = __builtin_amdgcn_wmma_f32_16x16x32_bf16(false, a, false, f1, (short)0, acc1, false, false);
        bf16x16 f2 = load_frag16(&bT[cur][bFragOff2], 8);
        acc2 = __builtin_amdgcn_wmma_f32_16x16x32_bf16(false, a, false, f2, (short)0, acc2, false, false);
        bf16x16 f3 = load_frag16(&bT[cur][bFragOff3], 8);
        acc3 = __builtin_amdgcn_wmma_f32_16x16x32_bf16(false, a, false, f3, (short)0, acc3, false, false);

        __syncthreads();   // all reads of buf[cur] done before it is refilled
    }
#else
    // fallback: direct-from-global fragments (round-1 path)
    const unsigned short* aPtr  = hsBf + (size_t)(mBase + laneLo) * Ktot + laneHi * 16;
    const unsigned short* bPtr0 = w + (size_t)(nBase +  0 + laneLo) * Ktot + laneHi * 16;
    const unsigned short* bPtr1 = w + (size_t)(nBase + 16 + laneLo) * Ktot + laneHi * 16;
    const unsigned short* bPtr2 = w + (size_t)(nBase + 32 + laneLo) * Ktot + laneHi * 16;
    const unsigned short* bPtr3 = w + (size_t)(nBase + 48 + laneLo) * Ktot + laneHi * 16;
    for (int kk = 0; kk < Ktot; kk += 32) {
        __builtin_prefetch(aPtr  + kk + 256, 0, 0);
        __builtin_prefetch(bPtr0 + kk + 256, 0, 0);
        bf16x16 a = load_frag16(aPtr + kk, 8);
        bf16x16 f0 = load_frag16(bPtr0 + kk, 8);
        acc0 = __builtin_amdgcn_wmma_f32_16x16x32_bf16(false, a, false, f0, (short)0, acc0, false, false);
        bf16x16 f1 = load_frag16(bPtr1 + kk, 8);
        acc1 = __builtin_amdgcn_wmma_f32_16x16x32_bf16(false, a, false, f1, (short)0, acc1, false, false);
        bf16x16 f2 = load_frag16(bPtr2 + kk, 8);
        acc2 = __builtin_amdgcn_wmma_f32_16x16x32_bf16(false, a, false, f2, (short)0, acc2, false, false);
        bf16x16 f3 = load_frag16(bPtr3 + kk, 8);
        acc3 = __builtin_amdgcn_wmma_f32_16x16x32_bf16(false, a, false, f3, (short)0, acc3, false, false);
    }
#endif

    // C/D layout: VGPR r -> row mBase + r + (lane>>4)*8, col nBase + j*16 + (lane&15)
    const int rowTop = mBase + laneHi * 8;
#define HRR_STORE_ACC(ACC, J)                                                   \
    do {                                                                        \
        int col  = nBase + (J) * 16 + laneLo;                                   \
        float bb = bias[col];                                                   \
        _Pragma("unroll")                                                       \
        for (int r = 0; r < 8; ++r)                                             \
            out[(size_t)(rowTop + r) * Ntot + col] = ACC[r] + bb;               \
    } while (0)
    HRR_STORE_ACC(acc0, 0);
    HRR_STORE_ACC(acc1, 1);
    HRR_STORE_ACC(acc2, 2);
    HRR_STORE_ACC(acc3, 3);
#undef HRR_STORE_ACC
}

// ---------------------------------------------------------------------------
// In-LDS radix-2 FFT, 2048 points, 256 threads
// ---------------------------------------------------------------------------
__device__ __forceinline__ void fft2048(float* re, float* im, int t, bool inverse) {
    for (int i = t; i < DD; i += 256) {
        int r = (int)(__brev((unsigned)i) >> (32 - LOGD));
        if (r > i) {
            float tr = re[i]; re[i] = re[r]; re[r] = tr;
            float ti = im[i]; im[i] = im[r]; im[r] = ti;
        }
    }
    __syncthreads();
    const float sgn = inverse ? 1.0f : -1.0f;
    for (int s = 1; s <= LOGD; ++s) {
        const int m = 1 << s, half = m >> 1;
        const float step = sgn * 6.28318530717958647692f / (float)m;
        for (int j = t; j < (DD >> 1); j += 256) {
            int grp = j >> (s - 1);
            int pos = j & (half - 1);
            int i1  = (grp << s) + pos;
            int i2  = i1 + half;
            float wi, wr;
            __sincosf(step * (float)pos, &wi, &wr);
            float br = re[i2], bi = im[i2];
            float xr = br * wr - bi * wi;
            float xi = br * wi + bi * wr;
            float ar = re[i1], ai = im[i1];
            re[i2] = ar - xr; im[i2] = ai - xi;
            re[i1] = ar + xr; im[i1] = ai + xi;
        }
        __syncthreads();
    }
}

// ---------------------------------------------------------------------------
// Forward: Ku,Vu unit-projection in freq domain, KV = Ku*Vu, raw Q spectrum.
// grid=(SS, BB), block=256
// ---------------------------------------------------------------------------
__global__ __launch_bounds__(256)
void hrr_fft_fwd(const float* __restrict__ qkv,
                 c32* __restrict__ kvF,
                 c32* __restrict__ qF) {
    __shared__ float reA[DD], imA[DD], reB[DD], imB[DD];
    const int s = blockIdx.x, b = blockIdx.y, t = threadIdx.x;
    const size_t MD  = (size_t)MM * DD;
    const size_t row = ((size_t)b * SS + s) * DD;
    const float* qrow = qkv + row;
    const float* krow = qkv + MD + row;
    const float* vrow = qkv + 2 * MD + row;

    for (int i = t; i < DD; i += 256) { reA[i] = krow[i]; imA[i] = 0.f; }
    __syncthreads();
    fft2048(reA, imA, t, false);
    for (int i = t; i < DD; i += 256) {
        float mag = sqrtf(reA[i] * reA[i] + imA[i] * imA[i]);
        float inv = 1.0f / (mag + 1e-8f);
        reA[i] *= inv; imA[i] *= inv;
    }
    for (int i = t; i < DD; i += 256) { reB[i] = vrow[i]; imB[i] = 0.f; }
    __syncthreads();
    fft2048(reB, imB, t, false);
    for (int i = t; i < DD; i += 256) {
        float mag = sqrtf(reB[i] * reB[i] + imB[i] * imB[i]);
        float inv = 1.0f / (mag + 1e-8f);
        float br = reB[i] * inv, bi = imB[i] * inv;
        c32 o;
        o.x = reA[i] * br - imA[i] * bi;
        o.y = reA[i] * bi + imA[i] * br;
        kvF[row + i] = o;
    }
    __syncthreads();
    for (int i = t; i < DD; i += 256) { reA[i] = qrow[i]; imA[i] = 0.f; }
    __syncthreads();
    fft2048(reA, imA, t, false);
    for (int i = t; i < DD; i += 256) {
        c32 o; o.x = reA[i]; o.y = imA[i];
        qF[row + i] = o;
    }
}

// ---------------------------------------------------------------------------
// Complex inclusive scan over S per (b, freq-bin), fused with conj(Q)*mem.
// In-place on kvF.  grid=(DD, BB), block=256, 16 elems/thread.
// ---------------------------------------------------------------------------
__global__ __launch_bounds__(256)
void hrr_scan_unbind(c32* __restrict__ kvF, const c32* __restrict__ qF) {
    const int f = blockIdx.x, b = blockIdx.y, t = threadIdx.x;
    const size_t base = (size_t)b * SS * DD + f;
    const int s0 = t * 16;

    c32 loc[16];
    c32 run; run.x = 0.f; run.y = 0.f;
#pragma unroll
    for (int i = 0; i < 16; ++i) {
        c32 v = kvF[base + (size_t)(s0 + i) * DD];
        run.x += v.x; run.y += v.y;
        loc[i] = run;
    }

    __shared__ c32 sums[256];
    sums[t] = run;
    __syncthreads();
    for (int off = 1; off < 256; off <<= 1) {
        c32 mine = sums[t];
        c32 add; add.x = 0.f; add.y = 0.f;
        if (t >= off) add = sums[t - off];
        __syncthreads();
        mine.x += add.x; mine.y += add.y;
        sums[t] = mine;
        __syncthreads();
    }
    c32 excl; excl.x = 0.f; excl.y = 0.f;
    if (t > 0) excl = sums[t - 1];

#pragma unroll
    for (int i = 0; i < 16; ++i) {
        size_t idx = base + (size_t)(s0 + i) * DD;
        c32 mem; mem.x = loc[i].x + excl.x; mem.y = loc[i].y + excl.y;
        c32 q = qF[idx];
        c32 o;                       // conj(q) * mem
        o.x = q.x * mem.x + q.y * mem.y;
        o.y = q.x * mem.y - q.y * mem.x;
        kvF[idx] = o;
    }
}

// ---------------------------------------------------------------------------
// Inverse FFT + epilogue: out = base + gate * Re(ifft)/N.  grid=(SS, BB)
// ---------------------------------------------------------------------------
__global__ __launch_bounds__(256)
void hrr_ifft_out(const c32* __restrict__ outF,
                  const float* __restrict__ baseOut,
                  const float* __restrict__ gate,
                  float* __restrict__ out) {
    __shared__ float re[DD], im[DD];
    const int s = blockIdx.x, b = blockIdx.y, t = threadIdx.x;
    const size_t row = ((size_t)b * SS + s) * DD;
    for (int i = t; i < DD; i += 256) {
        c32 v = outF[row + i];
        re[i] = v.x; im[i] = v.y;
    }
    __syncthreads();
    fft2048(re, im, t, true);
    const float g    = gate[0];
    const float invN = 1.0f / (float)DD;
    for (int i = t; i < DD; i += 256)
        out[row + i] = baseOut[row + i] + g * re[i] * invN;
}

// ---------------------------------------------------------------------------
// Launcher
// ---------------------------------------------------------------------------
extern "C" void kernel_launch(void* const* d_in, const int* in_sizes, int n_in,
                              void* d_out, int out_size, void* d_ws, size_t ws_size,
                              hipStream_t stream) {
    const float* hs   = (const float*)d_in[0];
    const float* base = (const float*)d_in[1];
    const float* Wq   = (const float*)d_in[2];
    const float* bq   = (const float*)d_in[3];
    const float* Wk   = (const float*)d_in[4];
    const float* bk   = (const float*)d_in[5];
    const float* Wv   = (const float*)d_in[6];
    const float* bv   = (const float*)d_in[7];
    const float* gate = (const float*)d_in[8];
    float* out = (float*)d_out;

    char* ws = (char*)d_ws;
    size_t off = 0;
    auto take = [&](size_t bytes) -> char* {
        char* p = ws + off;
        off += (bytes + 255) & ~(size_t)255;
        return p;
    };
    unsigned short* hsBf = (unsigned short*)take((size_t)MM * DD * 2);
    unsigned short* wBf  = (unsigned short*)take((size_t)3 * DD * DD * 2);
    float*          qkv  = (float*)take((size_t)3 * MM * DD * 4);
    c32*            kvF  = (c32*)take((size_t)BB * SS * DD * 8);
    c32*            qF   = (c32*)take((size_t)BB * SS * DD * 8);

    hrr_convert_bf16<<<2048, 256, 0, stream>>>(hs, Wq, Wk, Wv, hsBf, wBf);

    dim3 gGemm(DD / 256, MM / 32, 3);
    hrr_gemm_qkv<<<gGemm, 256, 0, stream>>>(hsBf, wBf, bq, bk, bv, qkv);

    hrr_fft_fwd<<<dim3(SS, BB), 256, 0, stream>>>(qkv, kvF, qF);
    hrr_scan_unbind<<<dim3(DD, BB), 256, 0, stream>>>(kvF, qF);
    hrr_ifft_out<<<dim3(SS, BB), 256, 0, stream>>>(kvF, base, gate, out);
}
